// FractionalOrderMaskGenerator_53584011985290
// MI455X (gfx1250) — matrix operand order repaired and checked by reading, not compiled, not run on hardware
//
#include <hip/hip_runtime.h>

typedef __attribute__((ext_vector_type(16))) __bf16 v16bf;
typedef __attribute__((ext_vector_type(8)))  float  v8f;

#define HH 192
#define WW 192
#define CC 64
#define EE 64
#define BB 4
#define HWP (HH*WW)      /* 36864  */
#define NPIX (BB*HWP)    /* 147456 */

__device__ __forceinline__ unsigned short f2bf(float f) {
  unsigned int u = __float_as_uint(f);
  u += 0x7FFFu + ((u >> 16) & 1u);          // round-to-nearest-even
  return (unsigned short)(u >> 16);
}
__device__ __forceinline__ float bf2f(unsigned short s) {
  return __uint_as_float(((unsigned int)s) << 16);
}

union Frag {
  uint4 q[2];
  v16bf v;
};

// ---------------------------------------------------------------------------
// Implicit-GEMM conv via WMMA bf16.  M = pixels (64/tile), N = 64 out ch,
// K = TAPS*TAPS*64 stepped by 32.  One WG = 256 thr = 8 waves; wave owns
// 16(M) x 32(N) -> two v8f accumulators.
// ---------------------------------------------------------------------------
template<int TAPS, int DIL, bool INF32>
__global__ void __launch_bounds__(256) conv_wmma_kernel(
    const void* __restrict__ in_, const unsigned short* __restrict__ wp,
    const float* __restrict__ biasEff, unsigned short* __restrict__ out)
{
  constexpr int CW = 64 + 2 * DIL;                 // staged x-width incl. halo
  __shared__ __align__(16) unsigned short sIn[TAPS][CW][72]; // 72 pad: b128-friendly

  const int bid = blockIdx.x;
  const int n   = bid / (HH * (WW / 64));
  const int r2  = bid % (HH * (WW / 64));
  const int y0  = r2 / (WW / 64);
  const int x0  = (r2 % (WW / 64)) * 64;
  const int tid = threadIdx.x;

  const float*          fin = (const float*)in_;
  const unsigned short* uin = (const unsigned short*)in_;

  constexpr int TOTAL = TAPS * CW * 64;
  for (int t = tid; t < TOTAL; t += 256) {
    int xi   = t % CW;
    int rest = t / CW;
    int r    = rest % TAPS;
    int ch   = rest / TAPS;
    int gy = y0 + (r - TAPS / 2) * DIL;
    int gx = x0 + xi - DIL;
    unsigned short val = 0;
    if (gy >= 0 && gy < HH && gx >= 0 && gx < WW) {
      long idx = ((long)(n * CC + ch) * HH + gy) * WW + gx;
      val = INF32 ? f2bf(fin[idx]) : uin[idx];
    }
    sIn[r][xi][ch] = val;
  }
  __syncthreads();

  const int lane = tid & 31;
  const int wid  = tid >> 5;
  const int mloc = (wid >> 1) * 16;   // pixel sub-tile
  const int n0   = (wid & 1) * 32;    // out-channel sub-tile
  const int lh   = lane >> 4;
  const int lm   = lane & 15;
  const int p    = mloc + lm;         // this lane's A row (pixel)

  v8f acc0 = {0.f,0.f,0.f,0.f,0.f,0.f,0.f,0.f};
  v8f acc1 = {0.f,0.f,0.f,0.f,0.f,0.f,0.f,0.f};

  #pragma unroll
  for (int tap = 0; tap < TAPS * TAPS; ++tap) {
    const int ky = tap / TAPS;
    const int kx = tap % TAPS;
    const int lx = p + kx * DIL;
    #pragma unroll
    for (int chalf = 0; chalf < 2; ++chalf) {
      // A: 16x32 bf16.  lane<16: e0-7 -> K0-7, e8-15 -> K16-23; lane>=16: +8.
      Frag a;
      const int b0 = chalf * 32 + lh * 8;
      a.q[0] = *(const uint4*)&sIn[ky][lx][b0];
      a.q[1] = *(const uint4*)&sIn[ky][lx][b0 + 16];
      // B: 32x16 bf16. col = lane&15, lane<16: e0-7 -> K0-7, e8-15 -> K16-23; lane>=16: +8.
      const int cb = chalf * 32 + lh * 8;
      Frag wb0, wb1;
      const unsigned short* wrow0 = wp + ((tap * 64 + (n0 + lm)) * 64 + cb);
      wb0.q[0] = ((const uint4*)wrow0)[0];
      wb0.q[1] = ((const uint4*)wrow0)[2];
      const unsigned short* wrow1 = wp + ((tap * 64 + (n0 + 16 + lm)) * 64 + cb);
      wb1.q[0] = ((const uint4*)wrow1)[0];
      wb1.q[1] = ((const uint4*)wrow1)[2];
      acc0 = __builtin_amdgcn_wmma_f32_16x16x32_bf16(false, a.v, false, wb0.v,
                                                     (short)0, acc0, false, false);
      acc1 = __builtin_amdgcn_wmma_f32_16x16x32_bf16(false, a.v, false, wb1.v,
                                                     (short)0, acc1, false, false);
    }
  }

  // D layout: VGPR v, lane<16 -> M=v, lane>=16 -> M=v+8; N = lane&15.
  const int och0 = n0 + lm;
  const int och1 = och0 + 16;
  const float bia0 = biasEff[och0];
  const float bia1 = biasEff[och1];
  #pragma unroll
  for (int v = 0; v < 8; ++v) {
    int pr = mloc + v + 8 * lh;
    int x  = x0 + pr;
    long base = ((long)(n * EE) * HH + y0) * WW + x;
    float v0 = acc0[v] + bia0; v0 = v0 > 0.f ? v0 : 0.f;
    float v1 = acc1[v] + bia1; v1 = v1 > 0.f ? v1 : 0.f;
    out[base + (long)och0 * HWP] = f2bf(v0);
    out[base + (long)och1 * HWP] = f2bf(v1);
  }
}

// ---------------------------------------------------------------------------
// Depthwise 3x3 (no activation), bf16 in -> bf16 out
// ---------------------------------------------------------------------------
__global__ void __launch_bounds__(256) dw_kernel(const unsigned short* __restrict__ in,
                                                 const float* __restrict__ wdw,
                                                 const float* __restrict__ bdw,
                                                 unsigned short* __restrict__ out)
{
  long gid = (long)blockIdx.x * 256 + threadIdx.x;
  if (gid >= (long)BB * EE * HWP) return;
  int x = (int)(gid % WW); long t = gid / WW;
  int y = (int)(t % HH);   t /= HH;
  int c = (int)(t % EE);   int n = (int)(t / EE);
  float acc = bdw[c];
  #pragma unroll
  for (int ky = 0; ky < 3; ++ky)
    #pragma unroll
    for (int kx = 0; kx < 3; ++kx) {
      int yy = y + ky - 1, xx = x + kx - 1;
      if (yy >= 0 && yy < HH && xx >= 0 && xx < WW)
        acc += bf2f(in[((long)(n * EE + c) * HH + yy) * WW + xx]) * wdw[c * 9 + ky * 3 + kx];
    }
  out[gid] = f2bf(acc);
}

// ---------------------------------------------------------------------------
// Attention: sigmoid(1x1(relu(1x1(x, wa1)), wa2))
// ---------------------------------------------------------------------------
__global__ void __launch_bounds__(256) att_kernel(const float* __restrict__ x,
                                                  const float* __restrict__ wa1,
                                                  const float* __restrict__ ba1,
                                                  const float* __restrict__ wa2,
                                                  const float* __restrict__ ba2,
                                                  float* __restrict__ att)
{
  __shared__ float sW1[16 * 64], sB1[16], sW2[16];
  for (int i = threadIdx.x; i < 1024; i += 256) sW1[i] = wa1[i];
  if (threadIdx.x < 16) { sB1[threadIdx.x] = ba1[threadIdx.x]; sW2[threadIdx.x] = wa2[threadIdx.x]; }
  __syncthreads();
  long pid = (long)blockIdx.x * 256 + threadIdx.x;
  if (pid >= NPIX) return;
  int n = (int)(pid / HWP); int rem = (int)(pid % HWP);
  float a[16];
  #pragma unroll
  for (int o = 0; o < 16; ++o) a[o] = sB1[o];
  for (int c = 0; c < 64; ++c) {
    float v = x[((long)(n * CC + c)) * HWP + rem];
    #pragma unroll
    for (int o = 0; o < 16; ++o) a[o] += v * sW1[o * 64 + c];
  }
  float z = ba2[0];
  #pragma unroll
  for (int o = 0; o < 16; ++o) { float r = a[o] > 0.f ? a[o] : 0.f; z += r * sW2[o]; }
  att[pid] = 1.f / (1.f + expf(-z));
}

// ---------------------------------------------------------------------------
// Weight packing: fold BN scale into weights, produce [tap][outCh][inCh] bf16
// ---------------------------------------------------------------------------
__global__ void pack33_kernel(const float* __restrict__ w, const float* __restrict__ b,
                              const float* __restrict__ g, const float* __restrict__ be,
                              unsigned short* __restrict__ wp, float* __restrict__ biasEff)
{
  int gid = blockIdx.x * 256 + threadIdx.x;
  const float inv = 1.f / sqrtf(1.f + 1e-5f);
  if (gid < 64) biasEff[gid] = b[gid] * (g[gid] * inv) + be[gid];
  if (gid < 9 * 64 * 64) {
    int tap = gid / 4096; int rem = gid % 4096;
    int e = rem / 64;     int c = rem % 64;
    float s = g[e] * inv;
    wp[(tap * 64 + e) * 64 + c] = f2bf(w[(e * 64 + c) * 9 + tap] * s);
  }
}

__global__ void packpw_kernel(const float* __restrict__ w, const float* __restrict__ b,
                              const float* __restrict__ g, const float* __restrict__ be,
                              unsigned short* __restrict__ wp, float* __restrict__ biasEff)
{
  int gid = blockIdx.x * 256 + threadIdx.x;
  const float inv = 1.f / sqrtf(1.f + 1e-5f);
  if (gid < 64) biasEff[gid] = b[gid] * (g[gid] * inv) + be[gid];
  if (gid < 64 * 64) {
    int e = gid / 64; int c = gid % 64;
    wp[e * 64 + c] = f2bf(w[e * 64 + c] * (g[e] * inv));
  }
}

// ---------------------------------------------------------------------------
// Direction-fusion matrix FW[25][24]: (pos-neg) mask per (dir, m)
// ---------------------------------------------------------------------------
__global__ void fw_kernel(float* __restrict__ FW)
{
  int idx = threadIdx.x;
  if (idx >= 600) return;
  const float s2 = 0.70710678f;
  const float vx[8] = {1.f,  s2, 0.f, -s2, -1.f, -s2, 0.f,  s2};
  const float vy[8] = {0.f,  s2, 1.f,  s2,  0.f, -s2, -1.f, -s2};
  int k = idx / 24; int col = idx % 24;
  int d = col / 3;  int m = col % 3;
  int i = k / 5, j = k % 5;
  float r0 = (float)(i - 2), r1 = (float)(j - 2);
  float proj = r0 * vy[d] + r1 * vx[d];
  int step = (int)rintf(proj);
  step = step < -2 ? -2 : (step > 2 ? 2 : step);
  FW[idx] = (step == m ? 1.f : 0.f) - (step == -m ? 1.f : 0.f);
}

// ---------------------------------------------------------------------------
// Final: 1x1 64->8 + tanh -> orders; analytic LUT interp; 25x24 fusion;
// attention scale; normalize over 25.
// ---------------------------------------------------------------------------
__global__ void __launch_bounds__(256) final_kernel(const unsigned short* __restrict__ h3,
                                                    const float* __restrict__ wf,
                                                    const float* __restrict__ bf,
                                                    const float* __restrict__ att,
                                                    const float* __restrict__ FW,
                                                    float* __restrict__ outMask,
                                                    float* __restrict__ outOrders)
{
  __shared__ float sFW[600], sWF[512], sBF[8];
  for (int i = threadIdx.x; i < 600; i += 256) sFW[i] = FW[i];
  for (int i = threadIdx.x; i < 512; i += 256) sWF[i] = wf[i];
  if (threadIdx.x < 8) sBF[threadIdx.x] = bf[threadIdx.x];
  __syncthreads();
  long pid = (long)blockIdx.x * 256 + threadIdx.x;
  if (pid >= NPIX) return;
  int n = (int)(pid / HWP); int rem = (int)(pid % HWP);

  float acc[8];
  #pragma unroll
  for (int d = 0; d < 8; ++d) acc[d] = sBF[d];
  for (int c = 0; c < 64; ++c) {
    float v = bf2f(h3[((long)(n * EE + c)) * HWP + rem]);
    #pragma unroll
    for (int d = 0; d < 8; ++d) acc[d] += v * sWF[d * 64 + c];
  }

  float coeff[24];
  #pragma unroll
  for (int d = 0; d < 8; ++d) {
    float order = 2.f * tanhf(acc[d]);
    outOrders[((long)(n * 8 + d)) * HWP + rem] = order;
    float t = (order + 2.f) * 0.25f;
    t = t < 0.f ? 0.f : (t > 1.f ? 1.f : t);
    float fidx = t * 199.f;
    float fi0 = floorf(fidx);
    float fi1 = ceilf(fidx);
    float wt1 = fidx - fi0;
    float wt0 = 1.f - wt1;
    float ov0 = -2.f + fi0 * (4.f / 199.f);   // LUT sample values (exact formulas)
    float ov1 = -2.f + fi1 * (4.f / 199.f);
    coeff[d * 3 + 0] = 1.f;
    coeff[d * 3 + 1] = -(ov0 * wt0 + ov1 * wt1);
    coeff[d * 3 + 2] = 0.5f * (ov0 * (ov0 - 1.f) * wt0 + ov1 * (ov1 - 1.f) * wt1);
  }

  float a = att[pid];
  float mask[25];
  float ssum = 0.f;
  #pragma unroll
  for (int k = 0; k < 25; ++k) {
    float f = 0.f;
    #pragma unroll
    for (int c = 0; c < 24; ++c) f += coeff[c] * sFW[k * 24 + c];
    float mk = f * 0.125f * a;
    mask[k] = mk;
    ssum += mk;
  }
  float invs = 1.f / (ssum + 1e-8f);
  #pragma unroll
  for (int k = 0; k < 25; ++k)
    outMask[((long)(n * 25 + k)) * HWP + rem] = mask[k] * invs;
}

// ---------------------------------------------------------------------------
extern "C" void kernel_launch(void* const* d_in, const int* in_sizes, int n_in,
                              void* d_out, int out_size, void* d_ws, size_t ws_size,
                              hipStream_t stream)
{
  (void)in_sizes; (void)n_in; (void)out_size; (void)ws_size;
  const float* fm  = (const float*)d_in[0];
  const float* w1  = (const float*)d_in[1];
  const float* b1  = (const float*)d_in[2];
  const float* g1  = (const float*)d_in[3];
  const float* be1 = (const float*)d_in[4];
  const float* w2  = (const float*)d_in[5];
  const float* b2  = (const float*)d_in[6];
  const float* g2  = (const float*)d_in[7];
  const float* be2 = (const float*)d_in[8];
  const float* wdw = (const float*)d_in[9];
  const float* bdw = (const float*)d_in[10];
  const float* wpw = (const float*)d_in[11];
  const float* bpw = (const float*)d_in[12];
  const float* g3  = (const float*)d_in[13];
  const float* be3 = (const float*)d_in[14];
  const float* wf  = (const float*)d_in[15];
  const float* bfv = (const float*)d_in[16];
  const float* wa1 = (const float*)d_in[17];
  const float* ba1 = (const float*)d_in[18];
  const float* wa2 = (const float*)d_in[19];
  const float* ba2 = (const float*)d_in[20];

  char* ws = (char*)d_ws;
  size_t off = 0;
  unsigned short* h1  = (unsigned short*)(ws + off); off += (size_t)BB * EE * HWP * 2; // 18.9MB
  unsigned short* h2  = (unsigned short*)(ws + off); off += (size_t)BB * EE * HWP * 2; // 18.9MB
  float*          att = (float*)(ws + off);          off += (size_t)NPIX * 4;          // 0.59MB
  unsigned short* wp1 = (unsigned short*)(ws + off); off += 9 * 64 * 64 * 2;
  unsigned short* wp2 = (unsigned short*)(ws + off); off += 9 * 64 * 64 * 2;
  unsigned short* wpp = (unsigned short*)(ws + off); off += 64 * 64 * 2 + 8192 - 64*64*2;
  float* bias1 = (float*)(ws + off); off += 256;
  float* bias2 = (float*)(ws + off); off += 256;
  float* biasp = (float*)(ws + off); off += 256;
  float* FW    = (float*)(ws + off); off += 2560;

  float* outMask   = (float*)d_out;
  float* outOrders = outMask + (size_t)BB * 25 * HWP;

  // weight prep + fusion matrix
  pack33_kernel<<<144, 256, 0, stream>>>(w1, b1, g1, be1, wp1, bias1);
  pack33_kernel<<<144, 256, 0, stream>>>(w2, b2, g2, be2, wp2, bias2);
  packpw_kernel<<<16, 256, 0, stream>>>(wpw, bpw, g3, be3, wpp, biasp);
  fw_kernel<<<1, 640, 0, stream>>>(FW);

  // attention branch (independent of conv stack)
  att_kernel<<<(NPIX + 255) / 256, 256, 0, stream>>>(fm, wa1, ba1, wa2, ba2, att);

  const int convGrid = BB * HH * (WW / 64);  // 2304
  // conv1 3x3 pad1 + BN + ReLU  (f32 in -> bf16 h1)
  conv_wmma_kernel<3, 1, true ><<<convGrid, 256, 0, stream>>>(fm, wp1, bias1, h1);
  // conv2 3x3 dil2 pad2 + BN + ReLU  (bf16 h1 -> bf16 h2)
  conv_wmma_kernel<3, 2, false><<<convGrid, 256, 0, stream>>>(h1, wp2, bias2, h2);
  // depthwise 3x3 (h2 -> h1 reuse)
  dw_kernel<<<(BB * EE * HWP + 255) / 256, 256, 0, stream>>>(h2, wdw, bdw, h1);
  // pointwise 1x1 + BN + ReLU  (h1 -> h2 reuse)
  conv_wmma_kernel<1, 0, false><<<convGrid, 256, 0, stream>>>(h1, wpp, biasp, h2);
  // final head: orders + adaptive mask
  final_kernel<<<(NPIX + 255) / 256, 256, 0, stream>>>(h2, wf, bfv, att, FW, outMask, outOrders);
}
